// SpatialAttention_89034672046706
// MI455X (gfx1250) — compile-verified, matrix operand order checked by reading, not run on hardware
//
#include <hip/hip_runtime.h>
#include <hip/hip_bf16.h>

typedef unsigned short u16;
typedef __bf16 v16bf __attribute__((ext_vector_type(16)));
typedef float  v8f   __attribute__((ext_vector_type(8)));
typedef unsigned int u32x4 __attribute__((ext_vector_type(4)));
typedef unsigned int u32x8 __attribute__((ext_vector_type(8)));
typedef int   i32x4v __attribute__((vector_size(16)));   // matches async-LDS builtin param
typedef float f32x4v __attribute__((ext_vector_type(4)));

// ---------- constants ----------
#define Bv   16
#define Cv   512
#define TNv  8192          // T*N = 128*64
#define NXv  67108864ULL   // B*C*T*N

// ---------- helpers ----------
__device__ __forceinline__ u16 f2bf(float f) {
    union { __bf16 h; u16 u; } v;
    v.h = (__bf16)f;                 // native v_cvt_*bf16_f32 on gfx1250
    return v.u;
}

// async global->LDS 16B copy (falls back to sync copy if builtin missing)
__device__ __forceinline__ void async_b128(const u16* g, u16* l) {
#if defined(__has_builtin) && __has_builtin(__builtin_amdgcn_global_load_async_to_lds_b128)
    __builtin_amdgcn_global_load_async_to_lds_b128(
        (i32x4v*)(void*)g, (i32x4v*)(void*)l, 0, 0);
#else
    *(uint4*)l = *(const uint4*)g;
#endif
}

template<int N>
__device__ __forceinline__ void wait_asynccnt() {
#if defined(__has_builtin) && __has_builtin(__builtin_amdgcn_s_wait_asynccnt)
    __builtin_amdgcn_s_wait_asynccnt(N);
#else
    asm volatile("s_wait_asynccnt %0" :: "i"(N) : "memory");
#endif
}

__device__ __forceinline__ void wait_tensorcnt0() {
#if defined(__has_builtin) && __has_builtin(__builtin_amdgcn_s_wait_tensorcnt)
    __builtin_amdgcn_s_wait_tensorcnt(0);
#else
    asm volatile("s_wait_tensorcnt 0x0" ::: "memory");
#endif
}

// TDM: 2-D bf16 tile (tileW x tileH elements, row stride strideEl) global -> LDS
__device__ __forceinline__ void tdm_load_2d_bf16(const u16* gsrc, void* lds_dst,
                                                 unsigned tileW, unsigned tileH,
                                                 unsigned strideEl, unsigned tensorW) {
    const unsigned ldsOff = (unsigned)(size_t)lds_dst;   // generic LDS ptr: low 32 bits = LDS offset
    const unsigned long long ga = (unsigned long long)(size_t)gsrc;
    u32x4 g0;
    g0[0] = 1u;                                          // count=1, user flags/gather = 0
    g0[1] = ldsOff;                                      // lds_addr [63:32]
    g0[2] = (unsigned)(ga & 0xFFFFFFFFu);                // global_addr[31:0]
    g0[3] = (unsigned)((ga >> 32) & 0x01FFFFFFu) | (2u << 30);  // global_addr[56:32], type=2
    u32x8 g1;
    g1[0] = 1u << 16;                                    // workgroup_mask=0, data_size=1 (2B)
    g1[1] = (tensorW & 0xFFFFu) << 16;                   // [47:32]=0, [63:48]=tensor_dim0[15:0]
    g1[2] = ((tensorW >> 16) & 0xFFFFu) | ((tileH & 0xFFFFu) << 16); // tensor_dim0[31:16], tensor_dim1[15:0]
    g1[3] = ((tileH >> 16) & 0xFFFFu) | (tileW << 16);   // tensor_dim1[31:16], tile_dim0
    g1[4] = tileH & 0xFFFFu;                             // tile_dim1, tile_dim2=0
    g1[5] = strideEl;                                    // tensor_dim0_stride[31:0]
    g1[6] = 0u;                                          // stride0[47:32]=0, stride1[15:0]=0
    g1[7] = 0u;
    asm volatile("tensor_load_to_lds %0, %1" :: "s"(g0), "s"(g1) : "memory");
}

// ---------- f32 -> bf16 convert (weights, no transpose) ----------
__global__ void cvt_bf16_kernel(const float* __restrict__ src, u16* __restrict__ dst, size_t n4) {
    size_t i = (size_t)blockIdx.x * blockDim.x + threadIdx.x;
    if (i >= n4) return;
    float4 v = ((const float4*)src)[i];
    uint2 pk;
    pk.x = (unsigned)f2bf(v.x) | ((unsigned)f2bf(v.y) << 16);
    pk.y = (unsigned)f2bf(v.z) | ((unsigned)f2bf(v.w) << 16);
    ((uint2*)dst)[i] = pk;
}

// ---------- x [b][c][tn] f32 -> xT [b][tn][c] bf16 (LDS-tiled transpose) ----------
__global__ void __launch_bounds__(256)
cvtT_kernel(const float* __restrict__ x, u16* __restrict__ xT) {
    const int b      = blockIdx.z;
    const int cBase  = blockIdx.y * 64;
    const int tnBase = blockIdx.x * 64;
    __shared__ __attribute__((aligned(32))) u16 Ts[64][72];   // +8 pad vs bank conflicts
    const int tid = threadIdx.x;
    {
        const int cl = tid >> 2, t0 = (tid & 3) * 16;
        const float* src = x + ((size_t)b * Cv + cBase + cl) * TNv + tnBase + t0;
#pragma unroll
        for (int j = 0; j < 16; j += 4) {
            float4 v = *(const float4*)(src + j);
            unsigned d0 = (unsigned)f2bf(v.x) | ((unsigned)f2bf(v.y) << 16);
            unsigned d1 = (unsigned)f2bf(v.z) | ((unsigned)f2bf(v.w) << 16);
            *(unsigned*)&Ts[cl][t0 + j]     = d0;
            *(unsigned*)&Ts[cl][t0 + j + 2] = d1;
        }
    }
    __syncthreads();
    {
        const int tr = tid >> 2, c0 = (tid & 3) * 16;
        u16* dst = xT + ((size_t)b * TNv + tnBase + tr) * Cv + cBase + c0;
        union { uint4 v[2]; u16 s[16]; } o;
#pragma unroll
        for (int j = 0; j < 16; ++j) o.s[j] = Ts[c0 + j][tr];
        *(uint4*)dst       = o.v[0];
        *(uint4*)(dst + 8) = o.v[1];
    }
}

// ---------- GEMM: out[b][m][col] = W[m][k] * XT[b][col][k] + bias[m] ----------
// 128x128 block tile, async double-buffered LDS staging, 8 waves x (2x4) WMMA accums
template<bool BF16OUT>
__global__ void __launch_bounds__(256)
gemm_bias_kernel(const u16* __restrict__ Wb, const u16* __restrict__ XT,
                 const float* __restrict__ bias,
                 u16* __restrict__ outB, float* __restrict__ outF,
                 int M, int K) {
    const int colBase = blockIdx.x * 128;
    const int mBase   = blockIdx.y * 128;
    const int b       = blockIdx.z;

    __shared__ __attribute__((aligned(32))) u16 sA[2][128 * 32];  // [m][k]
    __shared__ __attribute__((aligned(32))) u16 sB[2][128 * 32];  // [pixel][k]

    const int tid  = threadIdx.x;
    const int lane = tid & 31, wv = tid >> 5;
    const int wM = wv >> 1, wN = wv & 1;
    const int r  = lane & 15, hh = lane >> 4;

    const int row  = tid >> 1;          // 0..127
    const int koff = (tid & 1) * 16;    // 0 or 16
    const u16* gA = Wb + (size_t)(mBase + row) * K + koff;
    const u16* gB = XT + (size_t)b * (size_t)TNv * K + (size_t)(colBase + row) * K + koff;

    v8f acc[2][4] = {};
    const int nk = K >> 5;

    // prologue: stage tile 0 into buffer 0
    {
        async_b128(gA,     &sA[0][row * 32 + koff]);
        async_b128(gA + 8, &sA[0][row * 32 + koff + 8]);
        async_b128(gB,     &sB[0][row * 32 + koff]);
        async_b128(gB + 8, &sB[0][row * 32 + koff + 8]);
    }
    for (int kk = 0; kk < nk; ++kk) {
        const int cur = kk & 1;
        if (kk + 1 < nk) {   // stage next tile into the other buffer (overlaps WMMAs)
            const int kB = (kk + 1) * 32;
            async_b128(gA + kB,     &sA[cur ^ 1][row * 32 + koff]);
            async_b128(gA + kB + 8, &sA[cur ^ 1][row * 32 + koff + 8]);
            async_b128(gB + kB,     &sB[cur ^ 1][row * 32 + koff]);
            async_b128(gB + kB + 8, &sB[cur ^ 1][row * 32 + koff + 8]);
            wait_asynccnt<4>();   // current tile done, next 4 still in flight
        } else {
            wait_asynccnt<0>();
        }
        __syncthreads();
#pragma unroll
        for (int i = 0; i < 2; ++i) {
            v16bf a = *(const v16bf*)&sA[cur][(wM * 32 + i * 16 + r) * 32 + hh * 16];
#pragma unroll
            for (int j = 0; j < 4; ++j) {
                v16bf bb = *(const v16bf*)&sB[cur][(wN * 64 + j * 16 + r) * 32 + hh * 16];
                acc[i][j] = __builtin_amdgcn_wmma_f32_16x16x32_bf16(
                    false, a, false, bb, (short)0, acc[i][j], false, false);
            }
        }
        __syncthreads();   // all reads of buf[cur] done before it is re-staged
    }
    // epilogue: C layout row = (lane>>4)*8 + vgpr, col = lane&15
#pragma unroll
    for (int i = 0; i < 2; ++i) {
#pragma unroll
        for (int rr = 0; rr < 8; ++rr) {
            const int m = mBase + wM * 32 + i * 16 + hh * 8 + rr;
            const float bv = bias[m];
#pragma unroll
            for (int j = 0; j < 4; ++j) {
                const int col = colBase + wN * 64 + j * 16 + r;
                const float v = acc[i][j][rr] + bv;
                const size_t o = (size_t)b * (size_t)M * TNv + (size_t)m * TNv + col;
                if (BF16OUT) outB[o] = f2bf(v);
                else         outF[o] = v;
            }
        }
    }
}

// ---------- attention: S = (Q^T K)*scale ; O^T[n][d] = S[n][m] * V^T[m][d] ----------
__global__ void __launch_bounds__(128)
attn_kernel(const u16* __restrict__ qkv, u16* __restrict__ outT) {
    const int tIdx = blockIdx.x, h = blockIdx.y, b = blockIdx.z;
    __shared__ __attribute__((aligned(32))) u16 Qt[64 * 64]; // [n][d]
    __shared__ __attribute__((aligned(32))) u16 Kt[64 * 64]; // [m][d]
    __shared__ __attribute__((aligned(32))) u16 Vs[64 * 64]; // [d][m]
    __shared__ __attribute__((aligned(32))) u16 Ss[64 * 64]; // [n][m]

    const int tid = threadIdx.x;
    const size_t base = (size_t)b * 12582912ULL + (size_t)h * 64 * TNv + (size_t)tIdx * 64;

    // V tile via Tensor Data Mover (64x64 bf16, row stride TNv), one wave issues
    if (tid < 32)
        tdm_load_2d_bf16(qkv + base + (size_t)1024 * TNv, Vs, 64, 64, TNv, TNv);

    {   // transpose-load Q and K (pack d,d+1 pairs into dwords)
        const int p = tid >> 2, d0 = p * 2, n0 = (tid & 3) * 16;
        union Row { uint4 v[2]; u16 s[16]; };
        Row a0, a1;
        const u16* q0 = qkv + base + (size_t)d0 * TNv + n0;
        a0.v[0] = *(const uint4*)q0;         a0.v[1] = *(const uint4*)(q0 + 8);
        a1.v[0] = *(const uint4*)(q0 + TNv); a1.v[1] = *(const uint4*)(q0 + TNv + 8);
#pragma unroll
        for (int j = 0; j < 16; ++j)
            *(unsigned*)&Qt[(n0 + j) * 64 + d0] =
                (unsigned)a0.s[j] | ((unsigned)a1.s[j] << 16);
        const u16* k0 = q0 + (size_t)512 * TNv;
        a0.v[0] = *(const uint4*)k0;         a0.v[1] = *(const uint4*)(k0 + 8);
        a1.v[0] = *(const uint4*)(k0 + TNv); a1.v[1] = *(const uint4*)(k0 + TNv + 8);
#pragma unroll
        for (int j = 0; j < 16; ++j)
            *(unsigned*)&Kt[(n0 + j) * 64 + d0] =
                (unsigned)a0.s[j] | ((unsigned)a1.s[j] << 16);
    }
    if (tid < 32) wait_tensorcnt0();
    __syncthreads();

    const int lane = tid & 31, w = tid >> 5;
    const int r = lane & 15, hh = lane >> 4;

    // stage 1: S[n][m], wave w owns n-stripe [w*16, w*16+16)
    v8f acc1[4] = {};
#pragma unroll
    for (int kk = 0; kk < 2; ++kk) {
        v16bf a = *(const v16bf*)&Qt[(w * 16 + r) * 64 + kk * 32 + hh * 16];
#pragma unroll
        for (int j = 0; j < 4; ++j) {
            v16bf bb = *(const v16bf*)&Kt[(j * 16 + r) * 64 + kk * 32 + hh * 16];
            acc1[j] = __builtin_amdgcn_wmma_f32_16x16x32_bf16(
                false, a, false, bb, (short)0, acc1[j], false, false);
        }
    }
#pragma unroll
    for (int j = 0; j < 4; ++j)
#pragma unroll
        for (int rr = 0; rr < 8; ++rr) {
            const int n = w * 16 + hh * 8 + rr, m = j * 16 + r;
            Ss[n * 64 + m] = f2bf(acc1[j][rr] * 0.125f);   // 1/sqrt(64)
        }
    __syncthreads();

    // stage 2: O^T[n][d] = S[n][m] x V^T[m][d]; wave w owns n-stripe, tiles over d
    v8f acc2[4] = {};
#pragma unroll
    for (int kk = 0; kk < 2; ++kk) {
        v16bf a = *(const v16bf*)&Ss[(w * 16 + r) * 64 + kk * 32 + hh * 16];
#pragma unroll
        for (int j = 0; j < 4; ++j) {
            v16bf bb = *(const v16bf*)&Vs[(j * 16 + r) * 64 + kk * 32 + hh * 16];
            acc2[j] = __builtin_amdgcn_wmma_f32_16x16x32_bf16(
                false, a, false, bb, (short)0, acc2[j], false, false);
        }
    }
    // pixel-major output: outT[b][tn][c]
    u16* op = outT + (size_t)b * 4194304ULL + (size_t)(tIdx * 64) * Cv + (size_t)h * 64;
#pragma unroll
    for (int j = 0; j < 4; ++j)
#pragma unroll
        for (int rr = 0; rr < 8; ++rr) {
            const int n = w * 16 + hh * 8 + rr, d = j * 16 + r;
            op[(size_t)n * Cv + d] = f2bf(acc2[j][rr]);
        }
}

// ---------- BN stats / prep / apply ----------
__global__ void zero_f32_kernel(float* p, int n) {
    int i = blockIdx.x * blockDim.x + threadIdx.x;
    if (i < n) p[i] = 0.0f;
}

__global__ void __launch_bounds__(256)
stats_kernel(const float* __restrict__ projf, float* __restrict__ sums, float* __restrict__ sumsq) {
    const int c = blockIdx.x, b = blockIdx.y;
    const float4* p = (const float4*)(projf + ((size_t)b * Cv + c) * TNv);
    float s = 0.f, q = 0.f;
    for (int i = threadIdx.x; i < TNv / 4; i += 256) {
        float4 v = p[i];
        s += v.x + v.y + v.z + v.w;
        q += v.x * v.x + v.y * v.y + v.z * v.z + v.w * v.w;
    }
    __shared__ float rs[256], rq[256];
    rs[threadIdx.x] = s; rq[threadIdx.x] = q;
    __syncthreads();
    for (int off = 128; off > 0; off >>= 1) {
        if (threadIdx.x < off) {
            rs[threadIdx.x] += rs[threadIdx.x + off];
            rq[threadIdx.x] += rq[threadIdx.x + off];
        }
        __syncthreads();
    }
    if (threadIdx.x == 0) {
        atomicAdd(&sums[c],  rs[0]);
        atomicAdd(&sumsq[c], rq[0]);
    }
}

__global__ void bn_prep_kernel(const float* __restrict__ sums, const float* __restrict__ sumsq,
                               const float* __restrict__ gamma, const float* __restrict__ beta,
                               float* __restrict__ asc, float* __restrict__ bsh) {
    int c = blockIdx.x * blockDim.x + threadIdx.x;
    if (c >= Cv) return;
    const float inv_np = 1.0f / 131072.0f;       // B*T*N
    float mean = sums[c] * inv_np;
    float var  = sumsq[c] * inv_np - mean * mean;
    float sc   = gamma[c] * rsqrtf(var + 1e-5f);
    asc[c] = sc;
    bsh[c] = beta[c] - mean * sc;
}

__global__ void bn_apply_kernel(const float* __restrict__ projf, const float* __restrict__ x,
                                const float* __restrict__ asc, const float* __restrict__ bsh,
                                float* __restrict__ out) {
    size_t i4 = (size_t)blockIdx.x * blockDim.x + threadIdx.x;   // over NX/4
    const int c = (int)((i4 >> 11) & 511);
    f32x4v v  = ((const f32x4v*)projf)[i4];
    f32x4v xv = ((const f32x4v*)x)[i4];
    const float sc = asc[c], sh = bsh[c];
    f32x4v o = v * sc + sh + xv;
    __builtin_nontemporal_store(o, (f32x4v*)out + i4);   // write-only stream
}

// ---------- launcher ----------
extern "C" void kernel_launch(void* const* d_in, const int* in_sizes, int n_in,
                              void* d_out, int out_size, void* d_ws, size_t ws_size,
                              hipStream_t stream) {
    const float* x      = (const float*)d_in[0];
    const float* w_qkv  = (const float*)d_in[1];
    const float* b_qkv  = (const float*)d_in[2];
    const float* w_proj = (const float*)d_in[3];
    const float* b_proj = (const float*)d_in[4];
    const float* gamma  = (const float*)d_in[5];
    const float* beta   = (const float*)d_in[6];
    float* out = (float*)d_out;

    char* ws = (char*)d_ws;
    u16* xT     = (u16*)ws;  ws += NXv * 2;                 // x bf16, pixel-major [b][tn][c]
    u16* wqkvb  = (u16*)ws;  ws += (size_t)1536 * 512 * 2;
    u16* wprojb = (u16*)ws;  ws += (size_t)512 * 512 * 2;
    u16* qkvb   = (u16*)ws;  ws += 3 * NXv * 2;             // qkv bf16, channel-major [b][3C][tn]
    u16* attnT  = (u16*)ws;  ws += NXv * 2;                 // attn out bf16, pixel-major [b][tn][c]
    float* projf = (float*)ws; ws += NXv * 4;               // proj out f32, channel-major
    float* sums  = (float*)ws; ws += 512 * 4;
    float* sumsq = (float*)ws; ws += 512 * 4;
    float* asc   = (float*)ws; ws += 512 * 4;
    float* bsh   = (float*)ws; ws += 512 * 4;

    // 1. converts (x: transpose to pixel-major)
    cvtT_kernel<<<dim3(128, 8, 16), 256, 0, stream>>>(x, xT);
    cvt_bf16_kernel<<<(1536 * 512 / 4) / 256, 256, 0, stream>>>(w_qkv, wqkvb, 1536 * 512 / 4);
    cvt_bf16_kernel<<<(512 * 512 / 4) / 256, 256, 0, stream>>>(w_proj, wprojb, 512 * 512 / 4);
    // 2. QKV GEMM (bf16 out + bias)
    gemm_bias_kernel<true><<<dim3(64, 12, 16), 256, 0, stream>>>(
        wqkvb, xT, b_qkv, qkvb, nullptr, 1536, 512);
    // 3. attention per (b,h,t)
    attn_kernel<<<dim3(128, 8, 16), 128, 0, stream>>>(qkvb, attnT);
    // 4. proj GEMM (f32 out + bias)
    gemm_bias_kernel<false><<<dim3(64, 4, 16), 256, 0, stream>>>(
        wprojb, attnT, b_proj, nullptr, projf, 512, 512);
    // 5. BN stats -> fold -> apply (+ residual)
    zero_f32_kernel<<<4, 256, 0, stream>>>(sums, 1024);
    stats_kernel<<<dim3(512, 16), 256, 0, stream>>>(projf, sums, sumsq);
    bn_prep_kernel<<<2, 256, 0, stream>>>(sums, sumsq, gamma, beta, asc, bsh);
    bn_apply_kernel<<<(unsigned)(NXv / 4 / 256), 256, 0, stream>>>(projf, x, asc, bsh, out);
}
